// WrapCUBASpikingCNN_8847632629955
// MI455X (gfx1250) — matrix operand order repaired and checked by reading, not compiled, not run on hardware
//
#include <hip/hip_runtime.h>
#include <hip/hip_bf16.h>

typedef __attribute__((ext_vector_type(16))) _Float16 v16h;
typedef __attribute__((ext_vector_type(8)))  float    v8f;

#define NBLK   64
#define NTHR   256
#define TSTEPS 200

// ---------------- LIF update (CUBA): cur = .5*cur + psp; v = .75*v*(1-s) + cur; s = v > .5
__device__ inline float lif_step(float psp, float* cur, float* volt, float prevspk) {
    float c = 0.5f * (*cur) + psp;
    *cur = c;
    float v = 0.75f * (*volt) * (1.0f - prevspk) + c;
    *volt = v;
    return (v > 0.5f) ? 1.0f : 0.0f;
}

// ---------------- WMMA helpers (wave32, 16x16x32 f16 -> f32) ----------------
__device__ inline v8f wmma32(v16h a, v16h b, v8f c) {
    return __builtin_amdgcn_wmma_f32_16x16x32_f16(false, a, false, b, (short)0, c, false, false);
}

// B fragment: B[k][n] = W[n][k] with W row-major [N][K]. Lane: n = n0 + (lane&15),
// elems 0..7 -> K = k0 + (lane>>4)*8 + j; elems 8..15 -> K = k0 + 16 + (lane>>4)*8 + (j-8).
__device__ inline v16h loadB_W(const _Float16* __restrict__ W, int ldk, int n0, int k0, int lane) {
    const _Float16* p = W + (size_t)(n0 + (lane & 15)) * ldk + k0 + ((lane >> 4) << 3);
    v16h r;
#pragma unroll
    for (int j = 0; j < 8; ++j) { r[j] = p[j]; r[j + 8] = p[16 + j]; }
    return r;
}

// A fragment from row-major A[M][K]: lane m = m0 + (lane&15);
// elems 0..7  -> K = k0 + (lane>>4)*8 + j
// elems 8..15 -> K = k0 + 16 + (lane>>4)*8 + (j-8)
__device__ inline v16h loadA_rm(const _Float16* __restrict__ A, int ldk, int m0, int k0, int lane) {
    const _Float16* p = A + (size_t)(m0 + (lane & 15)) * ldk + k0 + ((lane >> 4) << 3);
    v16h r;
#pragma unroll
    for (int j = 0; j < 8; ++j) { r[j] = p[j]; r[j + 8] = p[16 + j]; }
    return r;
}

// conv2 im2col A: row m -> (b, y, x) of 6x6 output; col k = ci*9 + ky*3 + kx over s1[b][ci][8][8]
__device__ inline v16h loadA_conv2(const _Float16* __restrict__ s1, int m0, int k0, int lane) {
    int m = m0 + (lane & 15);
    int b = m / 36, s = m - b * 36;
    int y = s / 6, x = s - y * 6;
    int kb = k0 + ((lane >> 4) << 3);
    v16h r;
#pragma unroll
    for (int j = 0; j < 16; ++j) {
        int k = kb + (j & 7) + ((j >> 3) << 4);
        int ci = k / 9, rr = k - ci * 9;
        int ky = rr / 3, kx = rr - ky * 3;
        r[j] = s1[(((b << 6) + ci) * 8 + y + ky) * 8 + x + kx];
    }
    return r;
}

// conv3 A with AvgPool2d(2) fused: row m = batch b; col k = ci*9 + ky*3 + kx,
// value = mean of 2x2 window of s2[b][ci][6][6] at (2ky, 2kx)
__device__ inline v16h loadA_conv3(const _Float16* __restrict__ s2, int m0, int k0, int lane) {
    int b = m0 + (lane & 15);
    int kb = k0 + ((lane >> 4) << 3);
    v16h r;
#pragma unroll
    for (int j = 0; j < 16; ++j) {
        int k = kb + (j & 7) + ((j >> 3) << 4);
        int ci = k / 9, rr = k - ci * 9;
        int ky = rr / 3, kx = rr - ky * 3;
        const _Float16* p = s2 + (size_t)((b << 7) + ci) * 36 + (ky * 12) + (kx << 1);
        float v = (float)p[0] + (float)p[1] + (float)p[6] + (float)p[7];
        r[j] = (_Float16)(0.25f * v);
    }
    return r;
}

// ---------------- device-wide sense barrier ----------------
__device__ inline void grid_barrier(unsigned* cnt, unsigned* gen, unsigned nb) {
    __syncthreads();
    if (threadIdx.x == 0) {
        __threadfence();
        unsigned g = __hip_atomic_load(gen, __ATOMIC_ACQUIRE, __HIP_MEMORY_SCOPE_AGENT);
        unsigned arrived = __hip_atomic_fetch_add(cnt, 1u, __ATOMIC_ACQ_REL, __HIP_MEMORY_SCOPE_AGENT);
        if (arrived == nb - 1u) {
            __hip_atomic_store(cnt, 0u, __ATOMIC_RELAXED, __HIP_MEMORY_SCOPE_AGENT);
            __hip_atomic_store(gen, g + 1u, __ATOMIC_RELEASE, __HIP_MEMORY_SCOPE_AGENT);
        } else {
            while (__hip_atomic_load(gen, __ATOMIC_ACQUIRE, __HIP_MEMORY_SCOPE_AGENT) == g)
                __builtin_amdgcn_s_sleep(1);
        }
        __threadfence();
    }
    __syncthreads();
}

// ---------------- init: zero state / out / barrier, convert weights to fp16 ----------------
__global__ __launch_bounds__(NTHR) void snn_init(
        const float* __restrict__ w2, const float* __restrict__ w3, const float* __restrict__ tc,
        const float* __restrict__ rc, const float* __restrict__ f1,
        float* zf, _Float16* zh,
        _Float16* w2h, _Float16* w3h, _Float16* tch, _Float16* rech, _Float16* f1h,
        unsigned* bar, float* out) {
    int g = blockIdx.x * blockDim.x + threadIdx.x;
    int st = gridDim.x * blockDim.x;
    for (int i = g; i < 1228800; i += st) zf[i] = 0.f;
    for (int i = g; i < 663552;  i += st) zh[i] = (_Float16)0.f;
    for (int i = g; i < 73728;   i += st) w2h[i]  = (_Float16)w2[i];
    for (int i = g; i < 294912;  i += st) w3h[i]  = (_Float16)w3[i];
    for (int i = g; i < 196608;  i += st) tch[i]  = (_Float16)tc[i];
    for (int i = g; i < 65536;   i += st) rech[i] = (_Float16)rc[i];
    for (int i = g; i < 32768;   i += st) f1h[i]  = (_Float16)f1[i];
    for (int i = g; i < 128;     i += st) out[i]  = 0.f;
    if (g < 2) bar[g] = 0u;
}

// ---------------- persistent kernel: all 200 timesteps, 3 grid barriers each ----------------
__global__ __launch_bounds__(NTHR) void snn_persistent(
        const float* __restrict__ xin,                       // [64,1,10,10,200]
        const float* __restrict__ w1, const float* __restrict__ b1,
        const float* __restrict__ b2, const float* __restrict__ b3,
        const float* __restrict__ tcb, const float* __restrict__ recb,
        const float* __restrict__ f1b, const float* __restrict__ f2w,
        const float* __restrict__ tsw,
        float* cur1, float* volt1, float* cur2, float* volt2,
        float* cur3, float* volt3, float* curtc, float* volttc,
        float* curr_, float* voltr, float* curf, float* voltf,
        _Float16* s1, _Float16* s2, _Float16* ring, _Float16* stc,
        _Float16* spkr, _Float16* sf,
        const _Float16* __restrict__ w2h, const _Float16* __restrict__ w3h,
        const _Float16* __restrict__ tch, const _Float16* __restrict__ rech,
        const _Float16* __restrict__ f1h,
        unsigned* bar, float* out) {
    const int tid    = threadIdx.x;
    const int lane   = tid & 31;
    const int wv     = tid >> 5;
    const int gwave  = blockIdx.x * (NTHR / 32) + wv;
    const int nwaves = gridDim.x * (NTHR / 32);
    unsigned* cnt = bar;
    unsigned* gen = bar + 1;

#pragma unroll 1
    for (int t = 0; t < TSTEPS; ++t) {
        // ---- Phase A: block 0 runs tail stages for step t-1; blocks 1.. run conv1+LIF(t) ----
        if (blockIdx.x == 0) {
            if (t > 0) {
                const int tp = t - 1;
                // S5: temporal conv (3 taps over ring buffer) + LIF      [64 x 256]
                const int ntap = (tp >= 2) ? 3 : (tp + 1);
                for (int tl = wv; tl < 64; tl += 8) {
                    int m0 = (tl >> 4) << 4, n0 = (tl & 15) << 4;
                    v8f acc = {0.f,0.f,0.f,0.f,0.f,0.f,0.f,0.f};
                    for (int i = 0; i < ntap; ++i) {
                        const _Float16* A = ring + ((tp - i + 3) % 3) * 16384;
                        const _Float16* W = tch + i * 65536;
                        for (int k0 = 0; k0 < 256; k0 += 32)
                            acc = wmma32(loadA_rm(A, 256, m0, k0, lane), loadB_W(W, 256, n0, k0, lane), acc);
                    }
#pragma unroll
                    for (int r = 0; r < 8; ++r) {
                        int m = m0 + r + ((lane >> 4) << 3), n = n0 + (lane & 15);
                        float bias = tcb[n];
                        if (tp >= 1) bias += tcb[256 + n];
                        if (tp >= 2) bias += tcb[512 + n];
                        int idx = m * 256 + n;
                        float spike = lif_step(acc[r] + bias, &curtc[idx], &volttc[idx], (float)stc[idx]);
                        stc[idx] = (_Float16)spike;
                    }
                }
                __syncthreads();
                // S6: recurrent cell: psp = s_tc + prev_spk @ rec_w^T + rec_b; LIF (double-buffered)
                const _Float16* Ap = spkr + ((tp & 1) ^ 1) * 16384;
                _Float16* Sn = spkr + (tp & 1) * 16384;
                for (int tl = wv; tl < 64; tl += 8) {
                    int m0 = (tl >> 4) << 4, n0 = (tl & 15) << 4;
                    v8f acc = {0.f,0.f,0.f,0.f,0.f,0.f,0.f,0.f};
                    for (int k0 = 0; k0 < 256; k0 += 32)
                        acc = wmma32(loadA_rm(Ap, 256, m0, k0, lane), loadB_W(rech, 256, n0, k0, lane), acc);
#pragma unroll
                    for (int r = 0; r < 8; ++r) {
                        int m = m0 + r + ((lane >> 4) << 3), n = n0 + (lane & 15);
                        int idx = m * 256 + n;
                        float psp = acc[r] + (float)stc[idx] + recb[n];
                        float spike = lif_step(psp, &curr_[idx], &voltr[idx], (float)Ap[idx]);
                        Sn[idx] = (_Float16)spike;
                    }
                }
                __syncthreads();
                // S7: fc1 [64x128] = spk_r[64x256] @ fc1_w^T + b; LIF
                const _Float16* Ac = spkr + (tp & 1) * 16384;
                for (int tl = wv; tl < 32; tl += 8) {
                    int m0 = (tl >> 3) << 4, n0 = (tl & 7) << 4;
                    v8f acc = {0.f,0.f,0.f,0.f,0.f,0.f,0.f,0.f};
                    for (int k0 = 0; k0 < 256; k0 += 32)
                        acc = wmma32(loadA_rm(Ac, 256, m0, k0, lane), loadB_W(f1h, 256, n0, k0, lane), acc);
#pragma unroll
                    for (int r = 0; r < 8; ++r) {
                        int m = m0 + r + ((lane >> 4) << 3), n = n0 + (lane & 15);
                        int idx = m * 128 + n;
                        float spike = lif_step(acc[r] + f1b[n], &curf[idx], &voltf[idx], (float)sf[idx]);
                        sf[idx] = (_Float16)spike;
                    }
                }
                __syncthreads();
                // S8: fc2 + weighted accumulate into output
                if (tid < 128) {
                    int b = tid >> 1, o = tid & 1;
                    float s = 0.f;
#pragma unroll 8
                    for (int j = 0; j < 128; ++j) s += (float)sf[b * 128 + j] * f2w[o * 128 + j];
                    out[b * 2 + o] += s * tsw[tp];
                }
            }
        } else {
            // S1: conv1 (1->64ch, 3x3 on 10x10) + LIF, output spikes as fp16  [64,64,8,8]
            int g  = (blockIdx.x - 1) * NTHR + tid;
            int st = (gridDim.x - 1) * NTHR;
            for (int i = g; i < 262144; i += st) {
                int x = i & 7, y = (i >> 3) & 7, co = (i >> 6) & 63, b = i >> 12;
                float psp = b1[co];
                const float* wr = w1 + co * 9;
                const float* xb = xin + (size_t)b * 100 * TSTEPS + t;
#pragma unroll
                for (int ky = 0; ky < 3; ++ky)
#pragma unroll
                    for (int kx = 0; kx < 3; ++kx)
                        psp += wr[ky * 3 + kx] * xb[((y + ky) * 10 + (x + kx)) * TSTEPS];
                float spike = lif_step(psp, &cur1[i], &volt1[i], (float)s1[i]);
                s1[i] = (_Float16)spike;
            }
        }
        grid_barrier(cnt, gen, NBLK);

        // ---- Phase B: conv2 as im2col GEMM  M=2304 N=128 K=576, tile/wave + fused LIF ----
        for (int tl = gwave; tl < 1152; tl += nwaves) {
            int m0 = (tl >> 3) << 4, n0 = (tl & 7) << 4;
            v8f acc = {0.f,0.f,0.f,0.f,0.f,0.f,0.f,0.f};
            for (int k0 = 0; k0 < 576; k0 += 32) {
                if (k0 + 32 < 576)
                    __builtin_prefetch(w2h + (size_t)(n0 + (lane & 15)) * 576 + k0 + 32, 0, 1);
                acc = wmma32(loadA_conv2(s1, m0, k0, lane), loadB_W(w2h, 576, n0, k0, lane), acc);
            }
#pragma unroll
            for (int r = 0; r < 8; ++r) {
                int m = m0 + r + ((lane >> 4) << 3), n = n0 + (lane & 15);
                int b = m / 36, s = m - b * 36;
                int idx = ((b << 7) + n) * 36 + s;
                float spike = lif_step(acc[r] + b2[n], &cur2[idx], &volt2[idx], (float)s2[idx]);
                s2[idx] = (_Float16)spike;
            }
        }
        grid_barrier(cnt, gen, NBLK);

        // ---- Phase C: conv3 GEMM (pool fused into A-load) M=64 N=256 K=1152 + LIF -> ring ----
        for (int tl = gwave; tl < 64; tl += nwaves) {
            int m0 = (tl >> 4) << 4, n0 = (tl & 15) << 4;
            v8f acc = {0.f,0.f,0.f,0.f,0.f,0.f,0.f,0.f};
            for (int k0 = 0; k0 < 1152; k0 += 32)
                acc = wmma32(loadA_conv3(s2, m0, k0, lane), loadB_W(w3h, 1152, n0, k0, lane), acc);
            _Float16* rnew = ring + (t % 3) * 16384;
            const _Float16* rold = ring + ((t + 2) % 3) * 16384;
#pragma unroll
            for (int r = 0; r < 8; ++r) {
                int m = m0 + r + ((lane >> 4) << 3), n = n0 + (lane & 15);
                int idx = m * 256 + n;
                float spike = lif_step(acc[r] + b3[n], &cur3[idx], &volt3[idx], (float)rold[idx]);
                rnew[idx] = (_Float16)spike;
            }
        }
        grid_barrier(cnt, gen, NBLK);
    }

    // Final tail for step T-1
    if (blockIdx.x == 0) {
        const int tp = TSTEPS - 1;
        for (int tl = wv; tl < 64; tl += 8) {
            int m0 = (tl >> 4) << 4, n0 = (tl & 15) << 4;
            v8f acc = {0.f,0.f,0.f,0.f,0.f,0.f,0.f,0.f};
            for (int i = 0; i < 3; ++i) {
                const _Float16* A = ring + ((tp - i + 3) % 3) * 16384;
                const _Float16* W = tch + i * 65536;
                for (int k0 = 0; k0 < 256; k0 += 32)
                    acc = wmma32(loadA_rm(A, 256, m0, k0, lane), loadB_W(W, 256, n0, k0, lane), acc);
            }
#pragma unroll
            for (int r = 0; r < 8; ++r) {
                int m = m0 + r + ((lane >> 4) << 3), n = n0 + (lane & 15);
                float bias = tcb[n] + tcb[256 + n] + tcb[512 + n];
                int idx = m * 256 + n;
                float spike = lif_step(acc[r] + bias, &curtc[idx], &volttc[idx], (float)stc[idx]);
                stc[idx] = (_Float16)spike;
            }
        }
        __syncthreads();
        const _Float16* Ap = spkr + ((tp & 1) ^ 1) * 16384;
        _Float16* Sn = spkr + (tp & 1) * 16384;
        for (int tl = wv; tl < 64; tl += 8) {
            int m0 = (tl >> 4) << 4, n0 = (tl & 15) << 4;
            v8f acc = {0.f,0.f,0.f,0.f,0.f,0.f,0.f,0.f};
            for (int k0 = 0; k0 < 256; k0 += 32)
                acc = wmma32(loadA_rm(Ap, 256, m0, k0, lane), loadB_W(rech, 256, n0, k0, lane), acc);
#pragma unroll
            for (int r = 0; r < 8; ++r) {
                int m = m0 + r + ((lane >> 4) << 3), n = n0 + (lane & 15);
                int idx = m * 256 + n;
                float psp = acc[r] + (float)stc[idx] + recb[n];
                float spike = lif_step(psp, &curr_[idx], &voltr[idx], (float)Ap[idx]);
                Sn[idx] = (_Float16)spike;
            }
        }
        __syncthreads();
        const _Float16* Ac = spkr + (tp & 1) * 16384;
        for (int tl = wv; tl < 32; tl += 8) {
            int m0 = (tl >> 3) << 4, n0 = (tl & 7) << 4;
            v8f acc = {0.f,0.f,0.f,0.f,0.f,0.f,0.f,0.f};
            for (int k0 = 0; k0 < 256; k0 += 32)
                acc = wmma32(loadA_rm(Ac, 256, m0, k0, lane), loadB_W(f1h, 256, n0, k0, lane), acc);
#pragma unroll
            for (int r = 0; r < 8; ++r) {
                int m = m0 + r + ((lane >> 4) << 3), n = n0 + (lane & 15);
                int idx = m * 128 + n;
                float spike = lif_step(acc[r] + f1b[n], &curf[idx], &voltf[idx], (float)sf[idx]);
                sf[idx] = (_Float16)spike;
            }
        }
        __syncthreads();
        if (tid < 128) {
            int b = tid >> 1, o = tid & 1;
            float s = 0.f;
#pragma unroll 8
            for (int j = 0; j < 128; ++j) s += (float)sf[b * 128 + j] * f2w[o * 128 + j];
            out[b * 2 + o] += s * tsw[tp];
        }
    }
}

extern "C" void kernel_launch(void* const* d_in, const int* in_sizes, int n_in,
                              void* d_out, int out_size, void* d_ws, size_t ws_size,
                              hipStream_t stream) {
    const float* xin = (const float*)d_in[0];
    const float* w1  = (const float*)d_in[1];
    const float* b1  = (const float*)d_in[2];
    const float* w2  = (const float*)d_in[3];
    const float* b2  = (const float*)d_in[4];
    const float* w3  = (const float*)d_in[5];
    const float* b3  = (const float*)d_in[6];
    const float* tc  = (const float*)d_in[7];
    const float* tcb = (const float*)d_in[8];
    const float* rc  = (const float*)d_in[9];
    const float* rcb = (const float*)d_in[10];
    const float* f1  = (const float*)d_in[11];
    const float* f1b = (const float*)d_in[12];
    const float* f2w = (const float*)d_in[13];
    const float* tsw = (const float*)d_in[14];
    float* out = (float*)d_out;

    char* ws = (char*)d_ws;
    unsigned* bar = (unsigned*)ws;
    float* zf = (float*)(ws + 256);
    float* cur1  = zf;            float* volt1  = zf + 262144;
    float* cur2  = zf + 524288;   float* volt2  = zf + 819200;
    float* cur3  = zf + 1114112;  float* volt3  = zf + 1130496;
    float* curtc = zf + 1146880;  float* volttc = zf + 1163264;
    float* curr_ = zf + 1179648;  float* voltr  = zf + 1196032;
    float* curf  = zf + 1212416;  float* voltf  = zf + 1220608;
    _Float16* zh   = (_Float16*)(ws + 256 + 4915200);
    _Float16* s1   = zh;           _Float16* s2   = zh + 262144;
    _Float16* ring = zh + 557056;  _Float16* stc  = zh + 606208;
    _Float16* spkr = zh + 622592;  _Float16* sf   = zh + 655360;
    _Float16* wh   = zh + 663552;
    _Float16* w2h  = wh;           _Float16* w3h  = wh + 73728;
    _Float16* tch  = wh + 368640;  _Float16* rech = wh + 565248;
    _Float16* f1h  = wh + 630784;

    snn_init<<<512, NTHR, 0, stream>>>(w2, w3, tc, rc, f1, zf, zh,
                                       w2h, w3h, tch, rech, f1h, bar, out);
    snn_persistent<<<NBLK, NTHR, 0, stream>>>(
        xin, w1, b1, b2, b3, tcb, rcb, f1b, f2w, tsw,
        cur1, volt1, cur2, volt2, cur3, volt3, curtc, volttc,
        curr_, voltr, curf, voltf,
        s1, s2, ring, stc, spkr, sf,
        w2h, w3h, tch, rech, f1h, bar, out);
}
